// MultiHeadedAttention_55336358641758
// MI455X (gfx1250) — matrix operand order repaired and checked by reading, not compiled, not run on hardware
//
#include <hip/hip_runtime.h>
#include <hip/hip_bf16.h>
#include <stdint.h>

// ---------------------------------------------------------------------------
// MultiHeadedAttention for MI455X (gfx1250): bf16 WMMA everywhere.
// B=4, S=2048, D=1024, H=16, DK=64.
// ---------------------------------------------------------------------------

#define BATCH  4
#define SEQ    2048
#define DMODEL 1024
#define NHEAD  16
#define DHEAD  64

typedef __attribute__((ext_vector_type(16))) __bf16 v16bf;
typedef __attribute__((ext_vector_type(8)))  __bf16 v8bf;
typedef __attribute__((ext_vector_type(8)))  float  v8f;

union FragB { v16bf v; v8bf h[2]; };

static __device__ __forceinline__ __bf16 f2bf(float x) {
  // round-to-nearest-even f32 -> bf16
  union { float f; uint32_t u; } a; a.f = x;
  uint32_t r = a.u + 0x7FFFu + ((a.u >> 16) & 1u);
  union { uint16_t s; __bf16 b; } o; o.s = (uint16_t)(r >> 16);
  return o.b;
}

// ---------------------------------------------------------------------------
// fp32 -> bf16 conversion
// ---------------------------------------------------------------------------
__global__ void cvt_f32_to_bf16(const float* __restrict__ in,
                                __bf16* __restrict__ out, int n) {
  int i = blockIdx.x * blockDim.x + threadIdx.x;
  if (i < n) out[i] = f2bf(in[i]);
}

// ---------------------------------------------------------------------------
// V [B,S,D] (head-interleaved) -> Vt [B,H,DK,S]  (so PV B-fragments are
// contiguous along the kv dimension)
// ---------------------------------------------------------------------------
__global__ void transpose_heads_bf16(const __bf16* __restrict__ v,
                                     __bf16* __restrict__ vt) {
  size_t idx = (size_t)blockIdx.x * blockDim.x + threadIdx.x;
  const size_t total = (size_t)BATCH * NHEAD * DHEAD * SEQ;
  if (idx >= total) return;
  int s = (int)(idx % SEQ);
  size_t t = idx / SEQ;
  int d = (int)(t % DHEAD); t /= DHEAD;
  int h = (int)(t % NHEAD); t /= NHEAD;
  int b = (int)t;
  vt[idx] = v[((size_t)(b * SEQ + s)) * DMODEL + h * DHEAD + d];
}

// ---------------------------------------------------------------------------
// C[M,N] = A[M,K] @ W[N,K]^T + bias   (bf16 inputs, f32 accumulate)
// One wave computes a 16x64 output tile (4 WMMA n-tiles, A fragment reused 4x).
// A-fragment (16x32 bf16): lane L: row = L&15, k-halves selected by L>>4.
// B-fragment (32x16 bf16): lane L: col = L&15, k = (L>>4)*16 .. +15 contiguous.
// ---------------------------------------------------------------------------
template <typename OutT>
__global__ __launch_bounds__(256) void gemm_bf16_wmma(
    const __bf16* __restrict__ A,   // [M,K]
    const __bf16* __restrict__ W,   // [N,K]
    const float*  __restrict__ bias,// [N]
    OutT* __restrict__ C,           // [M,N]
    int M, int N, int K)
{
  const int lane = threadIdx.x & 31;
  const int wave = threadIdx.x >> 5;
  const int m16  = lane & 15;
  const int grp  = lane >> 4;

  const int tilesM  = M >> 4;
  const int tilesN4 = N >> 6;          // groups of 4 n-tiles (64 columns)
  const int job = blockIdx.x * 8 + wave;
  if (job >= tilesM * tilesN4) return; // wave-uniform: EXEC stays all-ones
  const int mt  = job % tilesM;
  const int nt4 = job / tilesM;

  const __bf16* aRow = A + (size_t)(mt * 16 + m16) * K;

  v8f acc[4];
  #pragma unroll
  for (int n = 0; n < 4; ++n) acc[n] = {};

  for (int k0 = 0; k0 < K; k0 += 32) {
    __builtin_prefetch(aRow + k0 + 256, 0, 1);   // global_prefetch_b8
    FragB af;
    af.h[0] = *(const v8bf*)(aRow + k0 + grp * 16);
    af.h[1] = *(const v8bf*)(aRow + k0 + grp * 16 + 8);
    #pragma unroll
    for (int n = 0; n < 4; ++n) {
      const int nCol = nt4 * 64 + n * 16 + m16;
      const __bf16* wRow = W + (size_t)nCol * K + k0 + grp * 16;
      FragB bfr;
      bfr.h[0] = *(const v8bf*)(wRow);
      bfr.h[1] = *(const v8bf*)(wRow + 8);
      acc[n] = __builtin_amdgcn_wmma_f32_16x16x32_bf16(
          false, af.v, false, bfr.v, (short)0, acc[n], false, false);
    }
  }

  #pragma unroll
  for (int n = 0; n < 4; ++n) {
    const int nCol = nt4 * 64 + n * 16 + m16;
    const float bb = bias[nCol];
    #pragma unroll
    for (int i = 0; i < 8; ++i) {
      const int row = mt * 16 + i + grp * 8;
      const float val = acc[n][i] + bb;
      if constexpr (sizeof(OutT) == 2) {
        ((__bf16*)C)[(size_t)row * N + nCol] = f2bf(val);
      } else {
        ((float*)C)[(size_t)row * N + nCol] = val;
      }
    }
  }
}

// ---------------------------------------------------------------------------
// Flash attention (causal), one wave per (b, h, 16-row q tile), kv tiles of 32.
// QK^T: 4 WMMAs / tile;  P*V: 4 WMMAs / tile.  P is staged through LDS to
// convert from C-layout to A-fragment layout (s_wait_dscnt + wave barrier).
// ---------------------------------------------------------------------------
__global__ __launch_bounds__(256) void flash_attn_bf16(
    const __bf16* __restrict__ Q,   // [B,S,D] head-interleaved
    const __bf16* __restrict__ Kt,  // [B,S,D] head-interleaved
    const __bf16* __restrict__ Vt,  // [B,H,DK,S]
    __bf16* __restrict__ Ctx)       // [B,S,D] head-interleaved
{
  __shared__ __attribute__((aligned(16))) __bf16 plds[8][16 * 32];

  const int lane = threadIdx.x & 31;
  const int wave = threadIdx.x >> 5;
  const int m16  = lane & 15;
  const int grp  = lane >> 4;

  const int qTiles = SEQ / 16;     // 128
  const int job = blockIdx.x * 8 + wave;
  if (job >= BATCH * NHEAD * qTiles) return;
  const int qt = job % qTiles;
  const int bh = job / qTiles;
  const int h  = bh % NHEAD;
  const int b  = bh / NHEAD;

  // Q fragment: 16x64 = two 16x32 A-fragments
  const __bf16* qPtr = Q + ((size_t)(b * SEQ + qt * 16 + m16)) * DMODEL + h * DHEAD;
  FragB qa[2];
  #pragma unroll
  for (int t = 0; t < 2; ++t) {
    qa[t].h[0] = *(const v8bf*)(qPtr + t * 32 + grp * 16);
    qa[t].h[1] = *(const v8bf*)(qPtr + t * 32 + grp * 16 + 8);
  }

  float mRun[8], lRun[8];
  v8f o[4];
  #pragma unroll
  for (int i = 0; i < 8; ++i) { mRun[i] = -1e30f; lRun[i] = 0.0f; }
  #pragma unroll
  for (int n = 0; n < 4; ++n) o[n] = {};

  const float scale = 0.125f;           // 1/sqrt(DK)
  const int nkv = ((qt * 16 + 16) + 31) >> 5;

  for (int j = 0; j < nkv; ++j) {
    const int kBase = j * 32;

    // ---- scores: S = Q @ K^T  (two 16-col halves, DK split into 2 k-steps)
    v8f s0 = {}, s1 = {};
    const __bf16* kPtr0 = Kt + ((size_t)(b * SEQ + kBase + m16)) * DMODEL + h * DHEAD;
    const __bf16* kPtr1 = Kt + ((size_t)(b * SEQ + kBase + 16 + m16)) * DMODEL + h * DHEAD;
    #pragma unroll
    for (int t = 0; t < 2; ++t) {
      FragB kb0, kb1;
      kb0.h[0] = *(const v8bf*)(kPtr0 + t * 32 + grp * 16);
      kb0.h[1] = *(const v8bf*)(kPtr0 + t * 32 + grp * 16 + 8);
      kb1.h[0] = *(const v8bf*)(kPtr1 + t * 32 + grp * 16);
      kb1.h[1] = *(const v8bf*)(kPtr1 + t * 32 + grp * 16 + 8);
      s0 = __builtin_amdgcn_wmma_f32_16x16x32_bf16(false, qa[t].v, false, kb0.v, (short)0, s0, false, false);
      s1 = __builtin_amdgcn_wmma_f32_16x16x32_bf16(false, qa[t].v, false, kb1.v, (short)0, s1, false, false);
    }

    // ---- online softmax over the 32 kv columns of this tile
    #pragma unroll
    for (int i = 0; i < 8; ++i) {
      const int row = qt * 16 + i + grp * 8;
      const int c0  = kBase + m16;
      float a0 = (c0 <= row)      ? s0[i] * scale : -1e30f;
      float a1 = (c0 + 16 <= row) ? s1[i] * scale : -1e30f;
      float mx = fmaxf(a0, a1);
      #pragma unroll
      for (int off = 1; off < 16; off <<= 1)
        mx = fmaxf(mx, __shfl_xor(mx, off, 32));
      const float mNew  = fmaxf(mRun[i], mx);
      const float alpha = __expf(mRun[i] - mNew);
      const float p0 = __expf(a0 - mNew);
      const float p1 = __expf(a1 - mNew);
      float rs = p0 + p1;
      #pragma unroll
      for (int off = 1; off < 16; off <<= 1)
        rs += __shfl_xor(rs, off, 32);
      lRun[i] = lRun[i] * alpha + rs;
      mRun[i] = mNew;
      #pragma unroll
      for (int n = 0; n < 4; ++n) o[n][i] *= alpha;
      // stage P (C-layout) into LDS, row-major 16x32
      plds[wave][(i + grp * 8) * 32 + m16]      = f2bf(p0);
      plds[wave][(i + grp * 8) * 32 + 16 + m16] = f2bf(p1);
    }

    asm volatile("s_wait_dscnt 0" ::: "memory");
    __builtin_amdgcn_wave_barrier();

    // re-load P in A-fragment layout (16x32)
    FragB pa;
    pa.h[0] = *(const v8bf*)(&plds[wave][m16 * 32 + grp * 16]);
    pa.h[1] = *(const v8bf*)(&plds[wave][m16 * 32 + grp * 16 + 8]);

    // ---- O += P @ V   (4 d-tiles of 16 columns)
    #pragma unroll
    for (int n = 0; n < 4; ++n) {
      const int dCol = n * 16 + m16;
      const __bf16* vPtr = Vt + ((size_t)((b * NHEAD + h) * DHEAD + dCol)) * SEQ
                              + kBase + grp * 16;
      FragB vb;
      vb.h[0] = *(const v8bf*)(vPtr);
      vb.h[1] = *(const v8bf*)(vPtr + 8);
      o[n] = __builtin_amdgcn_wmma_f32_16x16x32_bf16(
          false, pa.v, false, vb.v, (short)0, o[n], false, false);
    }

    asm volatile("s_wait_dscnt 0" ::: "memory");
    __builtin_amdgcn_wave_barrier();
  }

  // ---- normalize and write context (bf16, head-interleaved [B,S,D])
  #pragma unroll
  for (int i = 0; i < 8; ++i) {
    const float inv = 1.0f / lRun[i];
    const int row = qt * 16 + i + grp * 8;
    #pragma unroll
    for (int n = 0; n < 4; ++n) {
      Ctx[((size_t)(b * SEQ + row)) * DMODEL + h * DHEAD + n * 16 + m16] =
          f2bf(o[n][i] * inv);
    }
  }
}

// ---------------------------------------------------------------------------
// launch
// ---------------------------------------------------------------------------
extern "C" void kernel_launch(void* const* d_in, const int* in_sizes, int n_in,
                              void* d_out, int out_size, void* d_ws, size_t ws_size,
                              hipStream_t stream) {
  const float* query = (const float*)d_in[0];
  const float* key_  = (const float*)d_in[1];
  const float* value = (const float*)d_in[2];
  const float* Wq = (const float*)d_in[3];
  const float* bq = (const float*)d_in[4];
  const float* Wk = (const float*)d_in[5];
  const float* bk = (const float*)d_in[6];
  const float* Wv = (const float*)d_in[7];
  const float* bv = (const float*)d_in[8];
  const float* Wo = (const float*)d_in[9];
  const float* bo = (const float*)d_in[10];
  float* out = (float*)d_out;

  const size_t nTok = (size_t)BATCH * SEQ * DMODEL;  // 8M elements
  const size_t nW   = (size_t)DMODEL * DMODEL;       // 1M elements

  char* ws = (char*)d_ws;
  __bf16* bufA = (__bf16*)ws; ws += nTok * 2;  // input staging, later context
  __bf16* wbf  = (__bf16*)ws; ws += nW * 2;    // current weight (bf16)
  __bf16* qbf  = (__bf16*)ws; ws += nTok * 2;
  __bf16* kbf  = (__bf16*)ws; ws += nTok * 2;
  __bf16* vbf  = (__bf16*)ws; ws += nTok * 2;
  __bf16* vtbf = (__bf16*)ws; ws += nTok * 2;

  const int cvtBlk = 256;
  const int gTok = (int)((nTok + cvtBlk - 1) / cvtBlk);
  const int gW   = (int)((nW + cvtBlk - 1) / cvtBlk);

  const int M = BATCH * SEQ;                                  // 8192
  const int gemmBlocks = (M / 16) * (DMODEL / 64) / 8;        // 1024

  // Q = query @ Wq^T + bq
  cvt_f32_to_bf16<<<gTok, cvtBlk, 0, stream>>>(query, bufA, (int)nTok);
  cvt_f32_to_bf16<<<gW, cvtBlk, 0, stream>>>(Wq, wbf, (int)nW);
  gemm_bf16_wmma<__bf16><<<gemmBlocks, 256, 0, stream>>>(bufA, wbf, bq, qbf, M, DMODEL, DMODEL);
  // K
  cvt_f32_to_bf16<<<gTok, cvtBlk, 0, stream>>>(key_, bufA, (int)nTok);
  cvt_f32_to_bf16<<<gW, cvtBlk, 0, stream>>>(Wk, wbf, (int)nW);
  gemm_bf16_wmma<__bf16><<<gemmBlocks, 256, 0, stream>>>(bufA, wbf, bk, kbf, M, DMODEL, DMODEL);
  // V
  cvt_f32_to_bf16<<<gTok, cvtBlk, 0, stream>>>(value, bufA, (int)nTok);
  cvt_f32_to_bf16<<<gW, cvtBlk, 0, stream>>>(Wv, wbf, (int)nW);
  gemm_bf16_wmma<__bf16><<<gemmBlocks, 256, 0, stream>>>(bufA, wbf, bv, vbf, M, DMODEL, DMODEL);

  // V -> [B,H,DK,S]
  transpose_heads_bf16<<<(int)(nTok / 256), 256, 0, stream>>>(vbf, vtbf);

  // flash attention -> context (reuse bufA)
  flash_attn_bf16<<<(BATCH * NHEAD * (SEQ / 16)) / 8, 256, 0, stream>>>(qbf, kbf, vtbf, bufA);

  // out = context @ Wo^T + bo  (f32 output)
  cvt_f32_to_bf16<<<gW, cvtBlk, 0, stream>>>(Wo, wbf, (int)nW);
  gemm_bf16_wmma<float><<<gemmBlocks, 256, 0, stream>>>(bufA, wbf, bo, out, M, DMODEL, DMODEL);
}
